// Bert_coss_6047313953501
// MI455X (gfx1250) — matrix operand order repaired and checked by reading, not compiled, not run on hardware
//
#include <hip/hip_runtime.h>
#include <hip/hip_bf16.h>

typedef __attribute__((ext_vector_type(8)))  _Float16 v8h;
typedef __attribute__((ext_vector_type(16))) _Float16 v16h;
typedef __attribute__((ext_vector_type(8)))  float    v8f;

#define VEC 768
#define HD  256
#define BB  64
#define SS  512
#define TM  128
#define TN  128
#define KSTEP 32
#define LDA 40   // LDS row stride in halfs (32 + 8 pad -> conflict-free)

__device__ __forceinline__ v16h cat8(v8h lo, v8h hi) {
    return __builtin_shufflevector(lo, hi, 0,1,2,3,4,5,6,7,8,9,10,11,12,13,14,15);
}

// ---------------------------------------------------------------------------
// prep: convert mlp_w (fp32 -> f16) and zero the doc-sum accumulators.
// ---------------------------------------------------------------------------
__global__ __launch_bounds__(256) void prep_kernel(const float* __restrict__ W,
                                                   _Float16* __restrict__ Wh,
                                                   float* __restrict__ d1,
                                                   float* __restrict__ d2) {
    int g = blockIdx.x * 256 + threadIdx.x;
    if (g < HD * VEC) Wh[g] = (_Float16)W[g];
    if (g < BB * HD) { d1[g] = 0.f; d2[g] = 0.f; }
}

// ---------------------------------------------------------------------------
// Fused GEMM + bias + ReLU for both inputs (blockIdx.z selects X1/X2).
// Double-buffered LDS, one barrier per K-step, global prefetch overlaps WMMA.
// ---------------------------------------------------------------------------
__global__ __launch_bounds__(256) void gemm_relu_kernel(
    const float* __restrict__ X1, const float* __restrict__ X2,
    const _Float16* __restrict__ Wh, const float* __restrict__ bias,
    float* __restrict__ o1act,
    float* __restrict__ dsum1, float* __restrict__ dsum2)
{
    __shared__ _Float16 As[2][TM * LDA];
    __shared__ _Float16 Bs[2][TN * LDA];

    const int tid   = threadIdx.x;
    const int lane  = tid & 31;
    const int wave  = tid >> 5;
    const int waveM = wave & 1;   // 0..1  -> 64 rows each
    const int waveN = wave >> 1;  // 0..3  -> 32 cols each
    const int z     = blockIdx.z;
    const float* __restrict__ X = z ? X2 : X1;
    const int mBase = blockIdx.x * TM;
    const int nBase = blockIdx.y * TN;

    v8f c[4][2];
#pragma unroll
    for (int fm = 0; fm < 4; ++fm)
#pragma unroll
        for (int fn = 0; fn < 2; ++fn)
            c[fm][fn] = (v8f){0.f,0.f,0.f,0.f,0.f,0.f,0.f,0.f};

    // staging assignment: 256 threads cover 128 rows x 2 k-segments of 16
    const int ldRow = tid >> 1;
    const int ldK   = (tid & 1) * 16;
    const float*    gA = X  + (size_t)(mBase + ldRow) * VEC + ldK;
    const _Float16* gB = Wh + (size_t)(nBase + ldRow) * VEC + ldK;

    // per-lane fragment coordinates (ISA 16-bit A 16x32 / B 32x16 layouts)
    const int ar  = lane & 15;
    const int akb = (lane >> 4) * 8;
    const int bc  = lane & 15;
    const int bkb = (lane >> 4) * 8;

    // prefetch K-slab 0
    float4 fA0, fA1, fA2, fA3;
    v8h    pb0, pb1;
    {
        const float4* pA = (const float4*)gA;
        fA0 = pA[0]; fA1 = pA[1]; fA2 = pA[2]; fA3 = pA[3];
        const v8h* pB = (const v8h*)gB;
        pb0 = pB[0]; pb1 = pB[1];
    }

    const int NITER = VEC / KSTEP;   // 24
    for (int kk = 0; kk < NITER; ++kk) {
        const int buf = kk & 1;
        // ---- stage prefetched slab into LDS (A converted fp32 -> f16) ----
        v8h h0 = { (_Float16)fA0.x,(_Float16)fA0.y,(_Float16)fA0.z,(_Float16)fA0.w,
                   (_Float16)fA1.x,(_Float16)fA1.y,(_Float16)fA1.z,(_Float16)fA1.w };
        v8h h1 = { (_Float16)fA2.x,(_Float16)fA2.y,(_Float16)fA2.z,(_Float16)fA2.w,
                   (_Float16)fA3.x,(_Float16)fA3.y,(_Float16)fA3.z,(_Float16)fA3.w };
        _Float16* as = &As[buf][ldRow * LDA + ldK];
        *(v8h*)as       = h0;
        *(v8h*)(as + 8) = h1;
        _Float16* bst = &Bs[buf][ldRow * LDA + ldK];
        *(v8h*)bst       = pb0;
        *(v8h*)(bst + 8) = pb1;
        __syncthreads();

        // ---- prefetch next slab (overlaps the WMMAs below) ----
        if (kk + 1 < NITER) {
            const float4* pA = (const float4*)(gA + (kk + 1) * KSTEP);
            fA0 = pA[0]; fA1 = pA[1]; fA2 = pA[2]; fA3 = pA[3];
            const v8h* pB = (const v8h*)(gB + (kk + 1) * KSTEP);
            pb0 = pB[0]; pb1 = pB[1];
        }

        // ---- fragments + WMMA ----
        v16h a[4], b[2];
#pragma unroll
        for (int fm = 0; fm < 4; ++fm) {
            int row = waveM * 64 + fm * 16 + ar;
            v8h lo = *(const v8h*)&As[buf][row * LDA + akb];
            v8h hi = *(const v8h*)&As[buf][row * LDA + 16 + akb];
            a[fm] = cat8(lo, hi);
        }
#pragma unroll
        for (int fn = 0; fn < 2; ++fn) {
            int col = waveN * 32 + fn * 16 + bc;
            v8h lo = *(const v8h*)&Bs[buf][col * LDA + bkb];
            v8h hi = *(const v8h*)&Bs[buf][col * LDA + 16 + bkb];
            b[fn] = cat8(lo, hi);
        }
#pragma unroll
        for (int fm = 0; fm < 4; ++fm)
#pragma unroll
            for (int fn = 0; fn < 2; ++fn)
                c[fm][fn] = __builtin_amdgcn_wmma_f32_16x16x32_f16(
                    false, a[fm], false, b[fn], (short)0, c[fm][fn], false, false);
    }

    // ---- epilogue: doc sums (all z), then one uniform store branch ----
    const int batch = mBase >> 9;   // 128-row tiles lie within one batch
    float* __restrict__ dsum = z ? dsum2 : dsum1;
    const int colLo = nBase + waveN * 32 + (lane & 15);
    const float bs0 = bias[colLo];
    const float bs1 = bias[colLo + 16];

    float csum0 = 0.f, csum1 = 0.f;
#pragma unroll
    for (int fm = 0; fm < 4; ++fm)
#pragma unroll
        for (int i = 0; i < 8; ++i) {
            csum0 += fmaxf(c[fm][0][i] + bs0, 0.f);
            csum1 += fmaxf(c[fm][1][i] + bs1, 0.f);
        }
    csum0 += __shfl_xor(csum0, 16, 32);
    csum1 += __shfl_xor(csum1, 16, 32);
    if (lane < 16) {
        atomicAdd(&dsum[batch * HD + colLo],      csum0);
        atomicAdd(&dsum[batch * HD + colLo + 16], csum1);
    }

    if (z == 0) {
#pragma unroll
        for (int fn = 0; fn < 2; ++fn) {
            const float bsv = fn ? bs1 : bs0;
            const int col = colLo + fn * 16;
#pragma unroll
            for (int fm = 0; fm < 4; ++fm) {
                const int rowBase = mBase + waveM * 64 + fm * 16 + ((lane >> 4) << 3);
                float* __restrict__ p = o1act + (size_t)rowBase * HD + col;
#pragma unroll
                for (int i = 0; i < 8; ++i)
                    p[i * HD] = fmaxf(c[fm][fn][i] + bsv, 0.f);
            }
        }
    }
}

// ---------------------------------------------------------------------------
// head: doc means -> relu(fd) -> sigmoid(ff); writes output row 512.
// ---------------------------------------------------------------------------
__global__ __launch_bounds__(256) void head_kernel(
    const float* __restrict__ d1, const float* __restrict__ d2,
    const float* __restrict__ fd_w, const float* __restrict__ fd_b,
    const float* __restrict__ ff_w, const float* __restrict__ ff_b,
    float* __restrict__ out)
{
    __shared__ float cat[2 * HD];
    __shared__ float red[256];
    const int b = blockIdx.x, t = threadIdx.x;
    cat[t]      = d1[b * HD + t] * (1.f / (float)SS);
    cat[HD + t] = d2[b * HD + t] * (1.f / (float)SS);
    __syncthreads();
    float hh = fd_b[t];
    const float4* wrow = (const float4*)(fd_w + (size_t)t * (2 * HD));
    const float4* cv   = (const float4*)cat;
#pragma unroll 4
    for (int j = 0; j < (2 * HD) / 4; ++j) {
        float4 w = wrow[j], x = cv[j];
        hh += w.x * x.x + w.y * x.y + w.z * x.z + w.w * x.w;
    }
    hh = fmaxf(hh, 0.f);
    red[t] = hh * ff_w[t];
    __syncthreads();
    for (int off = 128; off > 0; off >>= 1) {
        if (t < off) red[t] += red[t + off];
        __syncthreads();
    }
    if (t == 0) {
        float x = red[0] + ff_b[0];
        out[SS * BB + b] = 1.f / (1.f + expf(-x));
    }
}

// ---------------------------------------------------------------------------
// attn: score[b,s] = o1e[b,s,:].o2_doc[b,:]; softmax over s (513 values,
// doc score in normalization, excluded from output); transposed store.
// ---------------------------------------------------------------------------
__global__ __launch_bounds__(512) void attn_kernel(
    const float* __restrict__ o1act,
    const float* __restrict__ d1, const float* __restrict__ d2,
    float* __restrict__ out)
{
    __shared__ float od1[HD], od2[HD];
    __shared__ float red[SS];
    __shared__ float sdocs;
    const int b = blockIdx.x, t = threadIdx.x;
    if (t < HD) {
        od1[t] = d1[b * HD + t] * (1.f / (float)SS);
        od2[t] = d2[b * HD + t] * (1.f / (float)SS);
    }
    __syncthreads();

    const float4* p = (const float4*)(o1act + ((size_t)b * SS + t) * HD);
    const float4* q = (const float4*)od2;
    float sc = 0.f;
#pragma unroll 4
    for (int j = 0; j < HD / 4; ++j) {
        float4 x = p[j], y = q[j];
        sc += x.x * y.x + x.y * y.y + x.z * y.z + x.w * y.w;
    }
    if (t == 0) {
        float sd = 0.f;
        for (int h = 0; h < HD; ++h) sd += od1[h] * od2[h];
        sdocs = sd;
    }
    red[t] = sc;
    __syncthreads();
    for (int off = 256; off > 0; off >>= 1) {
        if (t < off) red[t] = fmaxf(red[t], red[t + off]);
        __syncthreads();
    }
    const float mx = fmaxf(red[0], sdocs);
    __syncthreads();
    const float ex = expf(sc - mx);
    red[t] = ex;
    __syncthreads();
    for (int off = 256; off > 0; off >>= 1) {
        if (t < off) red[t] += red[t + off];
        __syncthreads();
    }
    const float denom = red[0] + expf(sdocs - mx);
    out[t * BB + b] = ex / denom;
}

// ---------------------------------------------------------------------------
extern "C" void kernel_launch(void* const* d_in, const int* in_sizes, int n_in,
                              void* d_out, int out_size, void* d_ws, size_t ws_size,
                              hipStream_t stream) {
    const float* X1    = (const float*)d_in[0];
    const float* X2    = (const float*)d_in[1];
    const float* mlp_w = (const float*)d_in[2];
    const float* mlp_b = (const float*)d_in[3];
    const float* fd_w  = (const float*)d_in[4];
    const float* fd_b  = (const float*)d_in[5];
    const float* ff_w  = (const float*)d_in[6];
    const float* ff_b  = (const float*)d_in[7];
    float* out = (float*)d_out;

    char* ws = (char*)d_ws;
    _Float16* Wh   = (_Float16*)ws;                              // 393,216 B
    float*    o1a  = (float*)(ws + (size_t)524288);              // 33,554,432 B
    float*    d1   = (float*)(ws + (size_t)524288 + 33554432);   // 65,536 B
    float*    d2   = d1 + BB * HD;                               // 65,536 B

    prep_kernel<<<768, 256, 0, stream>>>(mlp_w, Wh, d1, d2);
    gemm_relu_kernel<<<dim3((BB * SS) / TM, HD / TN, 2), 256, 0, stream>>>(
        X1, X2, Wh, mlp_b, o1a, d1, d2);
    head_kernel<<<BB, 256, 0, stream>>>(d1, d2, fd_w, fd_b, ff_w, ff_b, out);
    attn_kernel<<<BB, 512, 0, stream>>>(o1a, d1, d2, out);
}
